// TokenEmbeddingBranch2_1279900254218
// MI455X (gfx1250) — matrix operand order repaired and checked by reading, not compiled, not run on hardware
//
#include <hip/hip_runtime.h>
#include <hip/hip_bf16.h>
#include <math.h>

typedef __attribute__((ext_vector_type(16))) _Float16 v16h;
typedef __attribute__((ext_vector_type(8)))  _Float16 v8h;
typedef __attribute__((ext_vector_type(8)))  float    v8f;
typedef __attribute__((ext_vector_type(4)))  int      v4i;

#define B_   32
#define C_   512
#define F_   512
#define H_   4
#define DK_  128
#define MROWS (B_ * C_)          // 16384
#define MF    (MROWS * F_)       // 8388608 elements

// ---- CDNA5 async LDS-direct copy path ----
// Probe-confirmed signature (from hipcc diagnostic, round 2):
//   void __builtin_amdgcn_global_load_async_to_lds_b128(v4i AS1* src, v4i AS3* lds, imm offset, imm cpol)
#if __has_builtin(__builtin_amdgcn_global_load_async_to_lds_b128) && __has_builtin(__builtin_amdgcn_s_wait_asynccnt)
#define USE_ASYNC_LDS 1
typedef __attribute__((address_space(1))) v4i as1_v4i;   // global
typedef __attribute__((address_space(3))) v4i as3_v4i;   // LDS
#define ASYNC_CP16(gsrc, ldst) \
    __builtin_amdgcn_global_load_async_to_lds_b128((as1_v4i*)(gsrc), (as3_v4i*)(ldst), 0, 0)
#define ASYNC_WAIT0() __builtin_amdgcn_s_wait_asynccnt(0)
#else
#define USE_ASYNC_LDS 0
#define ASYNC_WAIT0()
#endif

static __device__ __forceinline__ v8h ldg8(const _Float16* p) {
    return *reinterpret_cast<const v8h*>(p);
}
static __device__ __forceinline__ v8f vzero8f() {
    v8f z;
#pragma unroll
    for (int i = 0; i < 8; ++i) z[i] = 0.0f;
    return z;
}
static __device__ __forceinline__ void cp16(const _Float16* gsrc, _Float16* ldst) {
#if USE_ASYNC_LDS
    ASYNC_CP16(gsrc, ldst);
#else
    *reinterpret_cast<v8h*>(ldst) = ldg8(gsrc);
#endif
}
static __device__ __forceinline__ void top5_insert(float (&top)[5], float v) {
    if (v > top[4]) {
        top[4] = v;
#pragma unroll
        for (int j = 4; j > 0; --j)
            if (top[j] > top[j - 1]) { float tm = top[j]; top[j] = top[j - 1]; top[j - 1] = tm; }
    }
}

// ---------------------------------------------------------------------------
// f32 -> f16 cast
// ---------------------------------------------------------------------------
__global__ void cast_f32_f16(const float* __restrict__ in, _Float16* __restrict__ out, int n) {
    int i = blockIdx.x * blockDim.x + threadIdx.x;
    if (i < n) out[i] = (_Float16)in[i];
}

// ---------------------------------------------------------------------------
// Y[M,N] = A[M,K] @ W[N,K]^T + bias ; f16 operands, f32 WMMA accumulate.
// Block tile 64x256, 8 waves (2x4), wave tile 32x64 = 2x4 wmma accumulators.
// Double-buffered LDS; next K-tile staged via async LDS-direct DMA (ASYNCcnt)
// while the current tile feeds the WMMA pipe. One barrier per K step.
// ---------------------------------------------------------------------------
__global__ __launch_bounds__(256)
void wmma_gemm_bias(const _Float16* __restrict__ A,
                    const _Float16* __restrict__ W,
                    const float* __restrict__ bias,
                    _Float16* __restrict__ outH,
                    float* __restrict__ outF,
                    int M, int N, int K)
{
    __shared__ _Float16 As[2][64][40];    // row stride 40 halves = 80B (16B aligned, conflict relief)
    __shared__ _Float16 Bs[2][256][40];

    const int m0 = blockIdx.y * 64;
    const int n0 = blockIdx.x * 256;
    const int t  = threadIdx.x;
    const int wave   = t >> 5;
    const int lane   = t & 31;
    const int laneLo = lane & 15;
    const int hiHalf = lane >> 4;      // 0 or 1
    const int wr = wave >> 2;          // 0..1 -> 32-row strip
    const int wc = wave & 3;           // 0..3 -> 64-col strip

    // staging indices: A tile 64x32 (1 seg/thread), B tile 256x32 (4 segs/thread = full row t)
    const int ar = t >> 2, ac = (t & 3) * 8;

    v8f acc[2][4];
#pragma unroll
    for (int i = 0; i < 2; ++i)
#pragma unroll
        for (int j = 0; j < 4; ++j) acc[i][j] = vzero8f();

    // prologue: stage K-tile 0 into buffer 0
    cp16(&A[(m0 + ar) * K + ac], &As[0][ar][ac]);
#pragma unroll
    for (int s = 0; s < 4; ++s)
        cp16(&W[(n0 + t) * K + s * 8], &Bs[0][t][s * 8]);
    ASYNC_WAIT0();
    __syncthreads();

    int buf = 0;
    for (int kk = 0; kk < K; kk += 32) {
        // issue DMA for the next K tile into the other buffer (overlaps WMMA below)
        if (kk + 32 < K) {
            cp16(&A[(m0 + ar) * K + kk + 32 + ac], &As[buf ^ 1][ar][ac]);
#pragma unroll
            for (int s = 0; s < 4; ++s)
                cp16(&W[(n0 + t) * K + kk + 32 + s * 8], &Bs[buf ^ 1][t][s * 8]);
        }

        // fragments from current buffer
        v16h afr[2], bfr[4];
#pragma unroll
        for (int mt = 0; mt < 2; ++mt) {
            const _Float16* ap = &As[buf][wr * 32 + mt * 16 + laneLo][0];
            reinterpret_cast<v8h*>(&afr[mt])[0] = *reinterpret_cast<const v8h*>(ap + hiHalf * 8);
            reinterpret_cast<v8h*>(&afr[mt])[1] = *reinterpret_cast<const v8h*>(ap + 16 + hiHalf * 8);
        }
#pragma unroll
        for (int nt = 0; nt < 4; ++nt) {
            const _Float16* bp = &Bs[buf][wc * 64 + nt * 16 + laneLo][0];
            reinterpret_cast<v8h*>(&bfr[nt])[0] = *reinterpret_cast<const v8h*>(bp + hiHalf * 8);
            reinterpret_cast<v8h*>(&bfr[nt])[1] = *reinterpret_cast<const v8h*>(bp + 16 + hiHalf * 8);
        }
#pragma unroll
        for (int mt = 0; mt < 2; ++mt)
#pragma unroll
            for (int nt = 0; nt < 4; ++nt)
                acc[mt][nt] = __builtin_amdgcn_wmma_f32_16x16x32_f16(
                    false, afr[mt], false, bfr[nt], (short)0, acc[mt][nt], false, false);

        ASYNC_WAIT0();        // this wave's DMA into buf^1 has landed
        __syncthreads();      // every wave done reading buf & done staging buf^1
        buf ^= 1;
    }

    // ---- epilogue: bias add + store (f16 and/or f32) ----
#pragma unroll
    for (int mt = 0; mt < 2; ++mt)
#pragma unroll
        for (int nt = 0; nt < 4; ++nt) {
            int col = n0 + wc * 64 + nt * 16 + laneLo;
            float bv = bias ? bias[col] : 0.0f;
#pragma unroll
            for (int i = 0; i < 8; ++i) {
                int row = m0 + wr * 32 + mt * 16 + hiHalf * 8 + i;
                float v = acc[mt][nt][i] + bv;
                if (outH) outH[row * N + col] = (_Float16)v;
                if (outF) outF[row * N + col] = v;
            }
        }
}

// ---------------------------------------------------------------------------
// Fused per-(b,h) attention on a 16-query-row tile:
//   S = Q K^T / sqrt(DK)   (WMMA, scores kept in LDS, never touch HBM)
//   parallel top-k(5) / softmax stats (chunked top-5 + merge; ds_add_f32 reduce)
//   O = P V                (WMMA, V chunk transposed through LDS)
// grid = B*H*(C/16) blocks of 256 threads (8 waves).
// ---------------------------------------------------------------------------
__global__ __launch_bounds__(256)
void attn_fused(const _Float16* __restrict__ Qb, const _Float16* __restrict__ Kb,
                const _Float16* __restrict__ Vb, _Float16* __restrict__ Ob,
                int use_topk)
{
    __shared__ float    S[16][512];     // 32 KB raw scores
    __shared__ _Float16 P[16][512];     // 16 KB probabilities (f16 for WMMA)
    __shared__ _Float16 Qs[16][128];    // 4 KB Q tile
    __shared__ _Float16 Vt[128][40];    // 10 KB transposed V chunk; reused as top-k scratch
    __shared__ float rowKth[16], rowMax[16], rowInv[16], rowSum[16];

    const int blk     = blockIdx.x;
    const int rowTile = blk & 31;            // C/16 = 32 tiles
    const int bh      = blk >> 5;
    const int h       = bh & (H_ - 1);
    const int b       = bh >> 2;             // H_ == 4
    const int c0      = rowTile * 16;

    const int t      = threadIdx.x;
    const int wave   = t >> 5;
    const int lane   = t & 31;
    const int laneLo = lane & 15;
    const int hiHalf = lane >> 4;

    const size_t base = ((size_t)b * C_) * F_ + (size_t)h * DK_;

    // ---- stage Q tile [16 x 128] (async LDS-direct when available) ----
    {
        int r = t >> 4, seg = (t & 15) * 8;
        cp16(&Qb[base + (size_t)(c0 + r) * F_ + seg], &Qs[r][seg]);
    }
    ASYNC_WAIT0();
    __syncthreads();

    // ---- scores: each wave owns 64 score columns (4 n-tiles) ----
    v8f sacc[4];
#pragma unroll
    for (int nt = 0; nt < 4; ++nt) sacc[nt] = vzero8f();
    const int nw = wave * 64;

    for (int ks = 0; ks < 4; ++ks) {           // DK = 4 * 32
        v16h a;
        const _Float16* ap = &Qs[laneLo][ks * 32];
        reinterpret_cast<v8h*>(&a)[0] = *reinterpret_cast<const v8h*>(ap + hiHalf * 8);
        reinterpret_cast<v8h*>(&a)[1] = *reinterpret_cast<const v8h*>(ap + 16 + hiHalf * 8);
#pragma unroll
        for (int nt = 0; nt < 4; ++nt) {
            int e = nw + nt * 16 + laneLo;     // key channel; K rows contiguous along k
            const _Float16* kp = &Kb[base + (size_t)e * F_ + ks * 32 + hiHalf * 8];
            v16h bf;
            reinterpret_cast<v8h*>(&bf)[0] = ldg8(kp);
            reinterpret_cast<v8h*>(&bf)[1] = ldg8(kp + 16);
            sacc[nt] = __builtin_amdgcn_wmma_f32_16x16x32_f16(
                false, a, false, bf, (short)0, sacc[nt], false, false);
        }
    }
    const float scale = 0.08838834764831845f;  // 1/sqrt(128)
#pragma unroll
    for (int nt = 0; nt < 4; ++nt)
#pragma unroll
        for (int i = 0; i < 8; ++i)
            S[hiHalf * 8 + i][nw + nt * 16 + laneLo] = sacc[nt][i] * scale;
    __syncthreads();

    // ---- phase A: per-chunk top-5 (16 threads per row x 32 entries each) ----
    float* scratch = reinterpret_cast<float*>(&Vt[0][0]);   // 16*16*5 floats = 5120B <= 10240B
    {
        int r = t & 15, ch = t >> 4;
        float top[5];
#pragma unroll
        for (int j = 0; j < 5; ++j) top[j] = -__builtin_inff();
        for (int j = 0; j < 32; ++j) top5_insert(top, S[r][ch * 32 + j]);
#pragma unroll
        for (int q = 0; q < 5; ++q) scratch[(r * 16 + ch) * 5 + q] = top[q];
    }
    __syncthreads();

    // ---- phase B: merge 16x5 candidates per row; softmax stats ----
    if (t < 16) {
        const int r = t;
        float top[5];
#pragma unroll
        for (int j = 0; j < 5; ++j) top[j] = -__builtin_inff();
        for (int ch = 0; ch < 16; ++ch)
#pragma unroll
            for (int q = 0; q < 5; ++q) top5_insert(top, scratch[(r * 16 + ch) * 5 + q]);
        float mx = top[0];
        rowMax[r] = mx;
        rowSum[r] = 0.0f;
        if (use_topk) {
            rowKth[r] = top[4];
            float d = 0.0f;
#pragma unroll
            for (int j = 0; j < 5; ++j) d += __expf(top[j] - mx);
            rowInv[r] = (d > 0.0f) ? 1.0f / d : 0.0f;   // nan_to_num path
        } else {
            rowKth[r] = -__builtin_inff();
        }
    }
    __syncthreads();

    // ---- dense path: parallel exp-sum reduction via LDS float atomics ----
    if (!use_topk) {
        int r = t & 15, ch = t >> 4;
        float mx = rowMax[r];
        float part = 0.0f;
        for (int j = 0; j < 32; ++j) part += __expf(S[r][ch * 32 + j] - mx);
        atomicAdd(&rowSum[r], part);               // ds_add_f32
        __syncthreads();
        if (t < 16) rowInv[t] = 1.0f / rowSum[t];
        __syncthreads();
    }

    // ---- probabilities -> f16 ----
    {
        int r = t & 15, ch = t >> 4;
        float kth = rowKth[r], mx = rowMax[r], inv = rowInv[r];
#pragma unroll 4
        for (int j = 0; j < 32; ++j) {
            int c = ch * 32 + j;
            float s = S[r][c];
            float p = (s >= kth) ? __expf(s - mx) * inv : 0.0f;
            P[r][c] = (_Float16)p;
        }
    }
    __syncthreads();

    // ---- O = P @ V : each wave owns one 16-wide slice of DK ----
    v8f oacc = vzero8f();
    for (int ks = 0; ks < 16; ++ks) {          // C = 16 * 32
        int k0 = ks * 32;
        // stage V chunk transposed: Vt[n][kk] = V[k0+kk][n]
        {
            int kkk = t >> 3;                  // 0..31
            int nb  = (t & 7) * 16;            // 0..112
            const _Float16* vp = &Vb[base + (size_t)(k0 + kkk) * F_ + nb];
            v8h v0 = ldg8(vp), v1 = ldg8(vp + 8);
#pragma unroll
            for (int j = 0; j < 8; ++j) {
                Vt[nb + j][kkk]     = v0[j];
                Vt[nb + 8 + j][kkk] = v1[j];
            }
        }
        __syncthreads();

        v16h a, bf;
        const _Float16* ap = &P[laneLo][k0];
        reinterpret_cast<v8h*>(&a)[0] = *reinterpret_cast<const v8h*>(ap + hiHalf * 8);
        reinterpret_cast<v8h*>(&a)[1] = *reinterpret_cast<const v8h*>(ap + 16 + hiHalf * 8);
        const _Float16* bp = &Vt[wave * 16 + laneLo][0];
        reinterpret_cast<v8h*>(&bf)[0] = *reinterpret_cast<const v8h*>(bp + hiHalf * 8);
        reinterpret_cast<v8h*>(&bf)[1] = *reinterpret_cast<const v8h*>(bp + 16 + hiHalf * 8);
        oacc = __builtin_amdgcn_wmma_f32_16x16x32_f16(
            false, a, false, bf, (short)0, oacc, false, false);
        __syncthreads();
    }

    // ---- store merged-head output tile ----
    {
        int d = wave * 16 + laneLo;
#pragma unroll
        for (int i = 0; i < 8; ++i) {
            int c = c0 + hiHalf * 8 + i;
            Ob[base + (size_t)c * F_ + d] = (_Float16)oacc[i];
        }
    }
}

// ---------------------------------------------------------------------------
// out = LayerNorm(x + glob) * gamma + beta ; one wave32 per 512-wide row.
// ---------------------------------------------------------------------------
__global__ __launch_bounds__(256)
void residual_layernorm(const float* __restrict__ x, const float* __restrict__ g,
                        const float* __restrict__ gamma, const float* __restrict__ beta,
                        float* __restrict__ out)
{
    const int wave = threadIdx.x >> 5, lane = threadIdx.x & 31;
    const int row  = blockIdx.x * 8 + wave;
    const float* xr = x + (size_t)row * F_;
    const float* gr = g + (size_t)row * F_;

    float hv[16];
    float s = 0.0f, s2 = 0.0f;
#pragma unroll
    for (int j = 0; j < 16; ++j) {
        int c = lane + j * 32;
        hv[j] = xr[c] + gr[c];
        s  += hv[j];
        s2 += hv[j] * hv[j];
    }
#pragma unroll
    for (int off = 16; off > 0; off >>= 1) {
        s  += __shfl_xor(s, off, 32);
        s2 += __shfl_xor(s2, off, 32);
    }
    const float mu  = s * (1.0f / F_);
    const float var = s2 * (1.0f / F_) - mu * mu;
    const float inv = rsqrtf(var + 1e-5f);
#pragma unroll
    for (int j = 0; j < 16; ++j) {
        int c = lane + j * 32;
        out[(size_t)row * F_ + c] = (hv[j] - mu) * inv * gamma[c] + beta[c];
    }
}

// ---------------------------------------------------------------------------
// Host-side orchestration (stream-only; graph-capture safe)
// ---------------------------------------------------------------------------
extern "C" void kernel_launch(void* const* d_in, const int* in_sizes, int n_in,
                              void* d_out, int out_size, void* d_ws, size_t ws_size,
                              hipStream_t stream)
{
    (void)in_sizes; (void)n_in; (void)out_size; (void)ws_size;

    const float* x = (const float*)d_in[0];
    // weight order: Wq1, Wk1, Wv1, Wo1, Wq2, Wk2, Wv2, Wo2
    const float* Wf[8]  = { (const float*)d_in[1],  (const float*)d_in[3],
                            (const float*)d_in[5],  (const float*)d_in[7],
                            (const float*)d_in[9],  (const float*)d_in[11],
                            (const float*)d_in[13], (const float*)d_in[15] };
    const float* bf_[8] = { (const float*)d_in[2],  (const float*)d_in[4],
                            (const float*)d_in[6],  (const float*)d_in[8],
                            (const float*)d_in[10], (const float*)d_in[12],
                            (const float*)d_in[14], (const float*)d_in[16] };
    const float* gamma = (const float*)d_in[17];
    const float* beta  = (const float*)d_in[18];

    char* ws = (char*)d_ws;
    _Float16* xh    = (_Float16*)(ws);                         // also reused for `local`
    _Float16* qh    = xh + (size_t)MF;
    _Float16* kh    = qh + (size_t)MF;
    _Float16* vh    = kh + (size_t)MF;
    _Float16* attnm = vh + (size_t)MF;
    float*    glob  = (float*)(ws + (size_t)5 * MF * 2);
    _Float16* wh    = (_Float16*)(ws + (size_t)5 * MF * 2 + (size_t)MF * 4);
    const int WSZ = F_ * F_;                                    // 262144 per weight

    // ---- one-time (per launch) f16 conversions ----
    cast_f32_f16<<<(MF + 255) / 256, 256, 0, stream>>>(x, xh, MF);
    for (int i = 0; i < 8; ++i)
        cast_f32_f16<<<(WSZ + 255) / 256, 256, 0, stream>>>(Wf[i], wh + (size_t)i * WSZ, WSZ);

    dim3 gemmGrid(F_ / 256, MROWS / 64);   // (2, 256)
    dim3 attnGrid(B_ * H_ * (C_ / 16));    // 4096

    // ---- stage 1: QKV, sparse local attention, Wo1 ----
    wmma_gemm_bias<<<gemmGrid, 256, 0, stream>>>(xh, wh + 0 * (size_t)WSZ, bf_[0], qh, nullptr, MROWS, F_, F_);
    wmma_gemm_bias<<<gemmGrid, 256, 0, stream>>>(xh, wh + 1 * (size_t)WSZ, bf_[1], kh, nullptr, MROWS, F_, F_);
    wmma_gemm_bias<<<gemmGrid, 256, 0, stream>>>(xh, wh + 2 * (size_t)WSZ, bf_[2], vh, nullptr, MROWS, F_, F_);
    attn_fused<<<attnGrid, 256, 0, stream>>>(qh, kh, vh, attnm, /*use_topk=*/1);
    wmma_gemm_bias<<<gemmGrid, 256, 0, stream>>>(attnm, wh + 3 * (size_t)WSZ, bf_[3], xh, nullptr, MROWS, F_, F_);

    // ---- stage 2: QKV on local output, dense global attention, Wo2 ----
    wmma_gemm_bias<<<gemmGrid, 256, 0, stream>>>(xh, wh + 4 * (size_t)WSZ, bf_[4], qh, nullptr, MROWS, F_, F_);
    wmma_gemm_bias<<<gemmGrid, 256, 0, stream>>>(xh, wh + 5 * (size_t)WSZ, bf_[5], kh, nullptr, MROWS, F_, F_);
    wmma_gemm_bias<<<gemmGrid, 256, 0, stream>>>(xh, wh + 6 * (size_t)WSZ, bf_[6], vh, nullptr, MROWS, F_, F_);
    attn_fused<<<attnGrid, 256, 0, stream>>>(qh, kh, vh, attnm, /*use_topk=*/0);
    wmma_gemm_bias<<<gemmGrid, 256, 0, stream>>>(attnm, wh + 7 * (size_t)WSZ, bf_[7], nullptr, glob, MROWS, F_, F_);

    // ---- residual + LayerNorm ----
    residual_layernorm<<<MROWS / 8, 256, 0, stream>>>(x, glob, gamma, beta, (float*)d_out);
}
